// SparseEncoder_45518063403086
// MI455X (gfx1250) — compile-verified, matrix operand order checked by reading, not run on hardware
//
#include <hip/hip_runtime.h>
#include <hip/hip_bf16.h>

typedef __attribute__((ext_vector_type(16))) _Float16     v16h;
typedef __attribute__((ext_vector_type(8)))  float        v8f;
typedef __attribute__((ext_vector_type(4)))  unsigned int v4u;
typedef __attribute__((ext_vector_type(2)))  float        v2f;

#define EPSBN 0.001f

// ---------------------------------------------------------------------------
// Implicit-GEMM 3D conv with fused BN+ReLU+mask epilogue, WMMA f32<-f16.
// One wave computes a 16(spatial-x) x 16(cout) output tile.
// K = ntaps * Cin. Tap decode/bounds/base hoisted per tap; fragment loads are
// 64/128-bit (activations NDHWC, weights pre-transposed to (tap, cout, cin)).
// Out-of-bounds / padded taps: A-fragment cndmasked to zero (A*B = 0), load
// addresses clamped in-range -> no divergent control flow in the hot loops.
// ---------------------------------------------------------------------------
struct ConvParams {
  const _Float16*      in;    // f16 input grid (NDHWC)
  const float*         in32;  // f32 input grid (layer 0 scatter result)
  const unsigned char* gate;  // output-resolution active mask
  const _Float16*      wts;   // weights (tap, cout, cin), f16
  const float *gamma, *beta, *mean, *var;
  _Float16* out;              // f16 output grid (NDHWC)
  float*    out32;            // final layer: fp32 transposed output
  int Din, Hin, Win;
  int Dout, Hout, Wout, Cout;
  int ntaps;
  int sz, sy, sx, pz, py, px;
};

union Frag {
  v16h h;
  unsigned int u[8];
  v4u  q[2];
};

template <int CS, int KH, int KW, bool IN32, bool OUT32>
__global__ __launch_bounds__(256) void conv3d_wmma_kernel(ConvParams p) {
  constexpr int CIN = 1 << CS;
  constexpr int KHW = KH * KW;

  const int lane  = threadIdx.x & 31;
  const int wave  = threadIdx.x >> 5;
  const int cout0 = wave << 4;
  const int x0    = blockIdx.x << 4;
  const int zo    = blockIdx.y / p.Hout;
  const int yo    = blockIdx.y - zo * p.Hout;
  const int half  = lane >> 4;
  const int mn    = lane & 15;   // A row (x-offset) == B column (cout-offset)

  const int xin0 = (x0 + mn) * p.sx - p.px;
  const int yin0 = yo * p.sy - p.py;
  const int zin0 = zo * p.sz - p.pz;
  const int cout = cout0 + mn;
  const long lanebase = ((long)zin0 * p.Hin + yin0) * p.Win + xin0;
  const int  hoff = half << 3;   // channel sub-offset for this lane half
  const v4u  z4 = {0u, 0u, 0u, 0u};

  v8f acc = {};

  if constexpr (CIN >= 32) {
    // One (Cin=32) or two (Cin=64) chunks per tap; all per-tap work hoisted.
    for (int tap = 0; tap < p.ntaps; ++tap) {
      const int kz  = tap / KHW;           // compile-time divisor
      const int rem = tap - kz * KHW;
      const int ky  = rem / KW;            // compile-time divisor
      const int kx  = rem - ky * KW;
      const int zi = zin0 + kz, yi = yin0 + ky, xi = xin0 + kx;
      const bool ok = (unsigned)zi < (unsigned)p.Din &&
                      (unsigned)yi < (unsigned)p.Hin &&
                      (unsigned)xi < (unsigned)p.Win;
      const long vox = ok ? ((lanebase + ((long)kz * p.Hin + ky) * p.Win + kx) << CS)
                          : 0;
      const _Float16* ab = p.in + vox + hoff;
      const _Float16* bb = p.wts + ((long)(tap * p.Cout + cout) << CS) + hoff;
#pragma unroll
      for (int c = 0; c < CIN / 32; ++c) {
        Frag A, B;
        const v4u r0 = *(const v4u*)(ab + (c << 5));
        const v4u r1 = *(const v4u*)(ab + (c << 5) + 16);
        A.q[0] = ok ? r0 : z4;
        A.q[1] = ok ? r1 : z4;
        B.q[0] = *(const v4u*)(bb + (c << 5));
        B.q[1] = *(const v4u*)(bb + (c << 5) + 16);
        acc = __builtin_amdgcn_wmma_f32_16x16x32_f16(false, A.h, false, B.h,
                                                     (short)0, acc, false, false);
      }
    }
  } else if constexpr (CIN == 16) {
    // One chunk covers two taps: fragment VGPRs 0-3 <- tap t0, 4-7 <- tap t0+1.
    for (int t0 = 0; t0 < p.ntaps; t0 += 2) {
      long voxs[2];
      bool oks[2];
      const _Float16* bb[2];
#pragma unroll
      for (int i = 0; i < 2; ++i) {
        const int tap = t0 + i;
        const bool tv = tap < p.ntaps;
        const int tc  = tv ? tap : t0;     // clamp for in-range addressing
        const int kz  = tc / KHW;
        const int rem = tc - kz * KHW;
        const int ky  = rem / KW;
        const int kx  = rem - ky * KW;
        const int zi = zin0 + kz, yi = yin0 + ky, xi = xin0 + kx;
        const bool ok = tv && (unsigned)zi < (unsigned)p.Din &&
                        (unsigned)yi < (unsigned)p.Hin &&
                        (unsigned)xi < (unsigned)p.Win;
        voxs[i] = ok ? ((lanebase + ((long)kz * p.Hin + ky) * p.Win + kx) << 4) : 0;
        oks[i]  = ok;
        bb[i]   = p.wts + ((long)(tc * p.Cout + cout) << 4) + hoff;
      }
      Frag A, B;
#pragma unroll
      for (int i = 0; i < 2; ++i) {
        const v4u ra = *(const v4u*)(p.in + voxs[i] + hoff);
        A.q[i] = oks[i] ? ra : z4;
        B.q[i] = *(const v4u*)bb[i];
      }
      acc = __builtin_amdgcn_wmma_f32_16x16x32_f16(false, A.h, false, B.h,
                                                   (short)0, acc, false, false);
    }
  } else {
    // CIN == 4 (layer 0, f32 source). Branch-free: clamp address, cndmask.
    const int nCh = ((p.ntaps << CS) + 31) >> 5;
    for (int kc = 0; kc < nCh; ++kc) {
      const int kg0 = kc << 5;
      Frag A, B;
#pragma unroll
      for (int v = 0; v < 8; ++v) {
        const int kb  = ((v & 4) << 2) | (half << 3) | ((v & 3) << 1);
        const int kg  = kg0 + kb;
        const int tap = kg >> CS;
        const int ci  = kg & (CIN - 1);
        const bool tv = tap < p.ntaps;
        const int tc  = tv ? tap : 0;      // clamp for in-range addressing
        const int kz  = tc / KHW;
        const int rem = tc - kz * KHW;
        const int ky  = rem / KW;
        const int kx  = rem - ky * KW;
        const int zi = zin0 + kz, yi = yin0 + ky, xi = xin0 + kx;
        const bool ok = tv && (unsigned)zi < (unsigned)p.Din &&
                        (unsigned)yi < (unsigned)p.Hin &&
                        (unsigned)xi < (unsigned)p.Win;
        const long vox =
            ok ? ((lanebase + ((long)kz * p.Hin + ky) * p.Win + kx) << CS) : 0;
        union { _Float16 h2[2]; unsigned int u; } t;
        if constexpr (IN32) {
          const v2f f2 = *(const v2f*)(p.in32 + vox + ci);   // b64 load
          t.h2[0] = (_Float16)f2.x;
          t.h2[1] = (_Float16)f2.y;
        } else {
          t.u = *(const unsigned int*)(p.in + vox + ci);
        }
        A.u[v] = ok ? t.u : 0u;
        B.u[v] = *(const unsigned int*)(p.wts + ((long)(tc * p.Cout + cout) << CS) + ci);
      }
      acc = __builtin_amdgcn_wmma_f32_16x16x32_f16(false, A.h, false, B.h,
                                                   (short)0, acc, false, false);
    }
  }

  // Fused BN + ReLU + mask gate epilogue. Each lane owns a single cout.
  const float sc = p.gamma[cout] * rsqrtf(p.var[cout] + EPSBN);
  const float bi = p.beta[cout];
  const float mu = p.mean[cout];
#pragma unroll
  for (int r = 0; r < 8; ++r) {
    const int m  = r + (half << 3);
    const int xo = x0 + m;
    if (xo < p.Wout) {
      const long pos = ((long)zo * p.Hout + yo) * p.Wout + xo;
      float val = 0.0f;
      if (p.gate[pos]) {
        val = (acc[r] - mu) * sc + bi;
        val = val > 0.0f ? val : 0.0f;
      }
      if constexpr (OUT32) {
        // transpose (d,h,w,c) -> (c*D + d, h, w)
        p.out32[(((long)cout * p.Dout + zo) * p.Hout + yo) * p.Wout + xo] = val;
      } else {
        p.out[pos * p.Cout + cout] = (_Float16)val;
      }
    }
  }
}

// ---------------------------------------------------------------------------
// Mask propagation for strided spconv: m_out = OR over taps of m_in.
// ---------------------------------------------------------------------------
struct MaskParams {
  const unsigned char* min;
  unsigned char* mout;
  int Din, Hin, Win, Dout, Hout, Wout;
  int kd, kh, kw, sz, sy, sx, pz, py, px;
};

__global__ void maskgen_kernel(MaskParams p) {
  const long i = (long)blockIdx.x * blockDim.x + threadIdx.x;
  const long tot = (long)p.Dout * p.Hout * p.Wout;
  if (i >= tot) return;
  const int xo = (int)(i % p.Wout);
  const long t = i / p.Wout;
  const int yo = (int)(t % p.Hout);
  const int zo = (int)(t / p.Hout);
  int any = 0;
  for (int kz = 0; kz < p.kd; ++kz)
    for (int ky = 0; ky < p.kh; ++ky)
      for (int kx = 0; kx < p.kw; ++kx) {
        const int zi = zo * p.sz - p.pz + kz;
        const int yi = yo * p.sy - p.py + ky;
        const int xi = xo * p.sx - p.px + kx;
        if ((unsigned)zi < (unsigned)p.Din && (unsigned)yi < (unsigned)p.Hin &&
            (unsigned)xi < (unsigned)p.Win)
          any |= p.min[((long)zi * p.Hin + yi) * p.Win + xi];
      }
  p.mout[i] = (unsigned char)(any ? 1 : 0);
}

// ---------------------------------------------------------------------------
// Voxel scatter-add into dense fp32 grid + mask set.
// ---------------------------------------------------------------------------
__global__ void scatter_kernel(const float* __restrict__ vf,
                               const int* __restrict__ coors,
                               float* grid, unsigned char* mask, int N) {
  const int i = blockIdx.x * blockDim.x + threadIdx.x;
  if (i >= N) return;
  const int z = coors[i * 4 + 1];
  const int y = coors[i * 4 + 2];
  const int x = coors[i * 4 + 3];
  const long cell = ((long)z * 200 + y) * 176 + x;
#pragma unroll
  for (int c = 0; c < 4; ++c)
    atomicAdd(&grid[cell * 4 + c], vf[i * 4 + c]);
  mask[cell] = 1;
}

// Weight convert + transpose: (tap, ci, co) f32 -> (tap, co, ci) f16
__global__ void cvtw_kernel(_Float16* dst, const float* __restrict__ src,
                            int Cin, int Cout, int total) {
  const int i = blockIdx.x * blockDim.x + threadIdx.x;
  if (i >= total) return;
  const int co = i % Cout;
  const int t  = i / Cout;
  const int ci = t % Cin;
  const int tap = t / Cin;
  dst[((long)tap * Cout + co) * Cin + ci] = (_Float16)src[i];
}

// ---------------------------------------------------------------------------
// Host side
// ---------------------------------------------------------------------------
static void run_conv(hipStream_t stream,
                     const _Float16* in, const float* in32,
                     int Din, int Hin, int Win, int Cin,
                     int Dout, int Hout, int Wout, int Cout,
                     int kd, int kh, int kw,
                     int sz, int sy, int sx,
                     int pz, int py, int px,
                     const unsigned char* gate, const _Float16* wts,
                     const float* g, const float* b, const float* mu, const float* var,
                     _Float16* out, float* out32) {
  ConvParams p;
  p.in = in; p.in32 = in32; p.gate = gate; p.wts = wts;
  p.gamma = g; p.beta = b; p.mean = mu; p.var = var;
  p.out = out; p.out32 = out32;
  p.Din = Din; p.Hin = Hin; p.Win = Win;
  p.Dout = Dout; p.Hout = Hout; p.Wout = Wout; p.Cout = Cout;
  p.ntaps = kd * kh * kw;
  p.sz = sz; p.sy = sy; p.sx = sx; p.pz = pz; p.py = py; p.px = px;
  dim3 grid((Wout + 15) / 16, (unsigned)(Dout * Hout));
  const unsigned blk = 32 * (Cout / 16);
  if (in32) {
    conv3d_wmma_kernel<2, 3, 3, true, false><<<grid, blk, 0, stream>>>(p);
  } else if (out32) {
    conv3d_wmma_kernel<6, 1, 1, false, true><<<grid, blk, 0, stream>>>(p);
  } else if (Cin == 16) {
    conv3d_wmma_kernel<4, 3, 3, false, false><<<grid, blk, 0, stream>>>(p);
  } else if (Cin == 32) {
    conv3d_wmma_kernel<5, 3, 3, false, false><<<grid, blk, 0, stream>>>(p);
  } else {
    conv3d_wmma_kernel<6, 3, 3, false, false><<<grid, blk, 0, stream>>>(p);
  }
}

static void run_mask(hipStream_t stream, const unsigned char* min, unsigned char* mout,
                     int Din, int Hin, int Win, int Dout, int Hout, int Wout,
                     int kd, int kh, int kw, int sz, int sy, int sx,
                     int pz, int py, int px) {
  MaskParams p;
  p.min = min; p.mout = mout;
  p.Din = Din; p.Hin = Hin; p.Win = Win;
  p.Dout = Dout; p.Hout = Hout; p.Wout = Wout;
  p.kd = kd; p.kh = kh; p.kw = kw;
  p.sz = sz; p.sy = sy; p.sx = sx; p.pz = pz; p.py = py; p.px = px;
  const long tot = (long)Dout * Hout * Wout;
  maskgen_kernel<<<(unsigned)((tot + 255) / 256), 256, 0, stream>>>(p);
}

extern "C" void kernel_launch(void* const* d_in, const int* in_sizes, int n_in,
                              void* d_out, int out_size, void* d_ws, size_t ws_size,
                              hipStream_t stream) {
  (void)n_in; (void)out_size; (void)ws_size;

  // d_in layout (setup_inputs dict order, params flattened leaf-wise):
  // 0 vf, 1 coors, 2 batch_size, then per conv block:
  // weight, gamma, beta, mean, var  (5 entries each)
  const int WI[12] = {3, 8, 13, 18, 23, 28, 33, 38, 43, 48, 53, 58};
  const int WCI[12] = {4, 16, 16, 32, 32, 32, 64, 64, 64, 64, 64, 64};
  const int WCO[12] = {16, 16, 32, 32, 32, 64, 64, 64, 64, 64, 64, 128};

  size_t off = 0;
  auto alloc = [&](size_t bytes) -> void* {
    void* p = (char*)d_ws + off;
    off += (bytes + 255) & ~(size_t)255;
    return p;
  };

  const size_t cells0 = (size_t)41 * 200 * 176;
  float*         grid0 = (float*)alloc(cells0 * 4 * sizeof(float));
  unsigned char* mask0 = (unsigned char*)alloc(cells0);
  _Float16*      bufA  = (_Float16*)alloc(cells0 * 16 * sizeof(_Float16));
  _Float16*      bufB  = (_Float16*)alloc(cells0 * 16 * sizeof(_Float16));
  unsigned char* m2    = (unsigned char*)alloc((size_t)21 * 100 * 88);
  unsigned char* m3    = (unsigned char*)alloc((size_t)11 * 50 * 44);
  unsigned char* m4    = (unsigned char*)alloc((size_t)5 * 25 * 22);
  unsigned char* mo    = (unsigned char*)alloc((size_t)2 * 25 * 22);
  _Float16* wf[12];
  for (int i = 0; i < 12; ++i)
    wf[i] = (_Float16*)alloc((size_t)in_sizes[WI[i]] * sizeof(_Float16));

  hipMemsetAsync(grid0, 0, cells0 * 4 * sizeof(float), stream);
  hipMemsetAsync(mask0, 0, cells0, stream);

  const int N = in_sizes[0] / 4;
  scatter_kernel<<<(N + 255) / 256, 256, 0, stream>>>(
      (const float*)d_in[0], (const int*)d_in[1], grid0, mask0, N);

  for (int i = 0; i < 12; ++i) {
    const int n = in_sizes[WI[i]];
    cvtw_kernel<<<(n + 255) / 256, 256, 0, stream>>>(
        wf[i], (const float*)d_in[WI[i]], WCI[i], WCO[i], n);
  }

  auto G = [&](int base, int k) { return (const float*)d_in[base + k]; };

  // ---- level 0 : (41,200,176) ----
  run_conv(stream, nullptr, grid0, 41, 200, 176, 4, 41, 200, 176, 16,
           3, 3, 3, 1, 1, 1, 1, 1, 1, mask0, wf[0],
           G(WI[0], 1), G(WI[0], 2), G(WI[0], 3), G(WI[0], 4), bufA, nullptr);
  run_conv(stream, bufA, nullptr, 41, 200, 176, 16, 41, 200, 176, 16,
           3, 3, 3, 1, 1, 1, 1, 1, 1, mask0, wf[1],
           G(WI[1], 1), G(WI[1], 2), G(WI[1], 3), G(WI[1], 4), bufB, nullptr);

  // ---- level 2 : (21,100,88) ----
  run_mask(stream, mask0, m2, 41, 200, 176, 21, 100, 88, 3, 3, 3, 2, 2, 2, 1, 1, 1);
  run_conv(stream, bufB, nullptr, 41, 200, 176, 16, 21, 100, 88, 32,
           3, 3, 3, 2, 2, 2, 1, 1, 1, m2, wf[2],
           G(WI[2], 1), G(WI[2], 2), G(WI[2], 3), G(WI[2], 4), bufA, nullptr);
  run_conv(stream, bufA, nullptr, 21, 100, 88, 32, 21, 100, 88, 32,
           3, 3, 3, 1, 1, 1, 1, 1, 1, m2, wf[3],
           G(WI[3], 1), G(WI[3], 2), G(WI[3], 3), G(WI[3], 4), bufB, nullptr);
  run_conv(stream, bufB, nullptr, 21, 100, 88, 32, 21, 100, 88, 32,
           3, 3, 3, 1, 1, 1, 1, 1, 1, m2, wf[4],
           G(WI[4], 1), G(WI[4], 2), G(WI[4], 3), G(WI[4], 4), bufA, nullptr);

  // ---- level 3 : (11,50,44) ----
  run_mask(stream, m2, m3, 21, 100, 88, 11, 50, 44, 3, 3, 3, 2, 2, 2, 1, 1, 1);
  run_conv(stream, bufA, nullptr, 21, 100, 88, 32, 11, 50, 44, 64,
           3, 3, 3, 2, 2, 2, 1, 1, 1, m3, wf[5],
           G(WI[5], 1), G(WI[5], 2), G(WI[5], 3), G(WI[5], 4), bufB, nullptr);
  run_conv(stream, bufB, nullptr, 11, 50, 44, 64, 11, 50, 44, 64,
           3, 3, 3, 1, 1, 1, 1, 1, 1, m3, wf[6],
           G(WI[6], 1), G(WI[6], 2), G(WI[6], 3), G(WI[6], 4), bufA, nullptr);
  run_conv(stream, bufA, nullptr, 11, 50, 44, 64, 11, 50, 44, 64,
           3, 3, 3, 1, 1, 1, 1, 1, 1, m3, wf[7],
           G(WI[7], 1), G(WI[7], 2), G(WI[7], 3), G(WI[7], 4), bufB, nullptr);

  // ---- level 4 : (5,25,22), pad (0,1,1) on the strided conv ----
  run_mask(stream, m3, m4, 11, 50, 44, 5, 25, 22, 3, 3, 3, 2, 2, 2, 0, 1, 1);
  run_conv(stream, bufB, nullptr, 11, 50, 44, 64, 5, 25, 22, 64,
           3, 3, 3, 2, 2, 2, 0, 1, 1, m4, wf[8],
           G(WI[8], 1), G(WI[8], 2), G(WI[8], 3), G(WI[8], 4), bufA, nullptr);
  run_conv(stream, bufA, nullptr, 5, 25, 22, 64, 5, 25, 22, 64,
           3, 3, 3, 1, 1, 1, 1, 1, 1, m4, wf[9],
           G(WI[9], 1), G(WI[9], 2), G(WI[9], 3), G(WI[9], 4), bufB, nullptr);
  run_conv(stream, bufB, nullptr, 5, 25, 22, 64, 5, 25, 22, 64,
           3, 3, 3, 1, 1, 1, 1, 1, 1, m4, wf[10],
           G(WI[10], 1), G(WI[10], 2), G(WI[10], 3), G(WI[10], 4), bufA, nullptr);

  // ---- output : (3,1,1) conv, stride (2,1,1), pad 0 -> (2,25,22,128), fp32
  run_mask(stream, m4, mo, 5, 25, 22, 2, 25, 22, 3, 1, 1, 2, 1, 1, 0, 0, 0);
  run_conv(stream, bufA, nullptr, 5, 25, 22, 64, 2, 25, 22, 128,
           3, 1, 1, 2, 1, 1, 0, 0, 0, mo, wf[11],
           G(WI[11], 1), G(WI[11], 2), G(WI[11], 3), G(WI[11], 4),
           nullptr, (float*)d_out);
}
